// RSNNNetwork_86870008528982
// MI455X (gfx1250) — matrix operand order repaired and checked by reading, not compiled, not run on hardware
//
#include <hip/hip_runtime.h>

// ---------------------------------------------------------------------------
// GLIFR RSNN, persistent single-WGP kernel for MI455X (gfx1250, wave32).
// Layer-1 GEMMs via v_wmma_f32_16x16x32_bf16 (f32 accumulate); GLIF state in
// VGPRs in the WMMA C-fragment layout; activation exchange via LDS; weights
// pre-converted to bf16 in workspace and re-streamed from WGP$/L2 each step
// via global_load_b128 saddr + 32-bit voffset (+imm) — anti-LICM asm keeps
// them out of registers; NT stores keep the 131MB spike stream out of L2.
// ---------------------------------------------------------------------------

typedef __attribute__((ext_vector_type(16))) __bf16 v16bf;
typedef __attribute__((ext_vector_type(8)))  __bf16 v8bf;
typedef __attribute__((ext_vector_type(8)))  float  v8f;

// global-memory (addrspace 1) views for the weight stream
typedef const __attribute__((address_space(1))) __bf16 gbf16;
typedef const __attribute__((address_space(1))) v8bf   gv8bf;

#define T_STEPS 1000
#define BATCH   32
#define N0      256
#define N1      512

// GLIF constants (DT=0.05, KM=0.2, R=1/9.43, KSYN=1, sigma_v=10, spike_r=20)
#define C_ISYN  0.95f                   // 1 - DT*KSYN
#define C_DTK   0.05f                   // DT*KSYN
#define C_VDEC  0.99f                   // 1 - DT*KM
#define C_VIN   (0.01f / 9.43f)         // DT*KM*R
#define C_A0D   0.99985f                // 1 - DT*0.003
#define C_A0M   (-0.459f)               // -9.18 * 1.0 * DT
#define C_A1D   0.995f                  // 1 - DT*0.1
#define C_A1M   (-9.947f)               // -198.94 * 1.0 * DT
#define THRESH  10.98612288668110f      // 10*ln(3)
#define INVSIG  0.1f
#define SPK_R   20.0f

// A-fragment (16x32 bf16, MxK) from LDS row-major activations:
// lane<16 holds K 0..15 of row M=lane; lane>=16 holds K 16..31.
__device__ __forceinline__ v16bf load_frag_a(const __bf16* rowp, int kc, int half) {
    union { v16bf v; v8bf h[2]; } u;
    const __bf16* p = rowp + kc * 32 + half * 16;
    u.h[0] = *(const v8bf*)(p);
    u.h[1] = *(const v8bf*)(p + 8);
    return u.v;
}

// B-fragment (32x16 bf16, KxN) from global row-major bf16 weights.
// voff already contains the whole divergent part (ncol*K + half*16) as a
// 32-bit unsigned element offset; only kc-dependent constants remain, so the
// backend can select global_load_b128 v, v_voff, s[base] offset:kc*64(+16).
__device__ __forceinline__ v16bf load_frag_b(gbf16* base, unsigned voff, unsigned kc) {
    union { v16bf v; v8bf h[2]; } u;
    gbf16* p = base + voff + kc * 32u;
    u.h[0] = *(gv8bf*)(p);
    u.h[1] = *(gv8bf*)(p + 8);
    return u.v;
}

__device__ __forceinline__ float glif_rate(float V) {
    // spike_r * sigmoid((V - thresh)/sigma_v)
    return SPK_R / (1.0f + __expf((THRESH - V) * INVSIG));
}

// ---- one-shot f32 -> bf16 weight conversion into workspace -----------------
__global__ void convert_w(const float* __restrict__ W1, const float* __restrict__ Wr,
                          __bf16* __restrict__ W1b, __bf16* __restrict__ Wrb) {
    int i = blockIdx.x * blockDim.x + threadIdx.x;
    if (i < N1 * N0) W1b[i] = (__bf16)W1[i];
    if (i < N1 * N1) Wrb[i] = (__bf16)Wr[i];
}

// ---- persistent recurrent kernel: 1 block, 32 waves on one WGP -------------
__global__ __launch_bounds__(1024) void rsnn_persistent(
    const float*  __restrict__ x,      // [T,B,1,N0]
    const __bf16* __restrict__ W1b,    // [N1,N0] row-major
    const __bf16* __restrict__ Wrb,    // [N1,N1] row-major
    float*        __restrict__ out)    // [2T,B,1,N1]
{
    __shared__ __align__(16) __bf16 o0[BATCH * N0];  // delayed layer-0 rates
    __shared__ __align__(16) __bf16 o1[BATCH * N1];  // delayed layer-1 rates

    const int tid  = threadIdx.x;
    const int lane = tid & 31;
    const int wave = tid >> 5;            // N-tile 0..31 (16 cols each)
    const int half = (lane >> 4) & 1;
    const int l16  = lane & 15;
    const int ncol = wave * 16 + l16;     // this lane's output column
    // one 32-bit divergent element offset per weight matrix (incl. half*16)
    const unsigned voff1 = (unsigned)(ncol * N0 + half * 16);
    const unsigned voffr = (unsigned)(ncol * N1 + half * 16);

    for (int i = tid; i < BATCH * N0; i += 1024) o0[i] = (__bf16)0.0f;
    for (int i = tid; i < BATCH * N1; i += 1024) o1[i] = (__bf16)0.0f;

    // layer-0 state: 8 elements/thread, idx = tid + 1024*k
    float V0[8], A00[8], A01[8], I0[8];
#pragma unroll
    for (int k = 0; k < 8; ++k) { V0[k]=0.f; A00[k]=0.f; A01[k]=0.f; I0[k]=0.f; }

    // layer-1 state: 16 elements/thread in C-fragment layout:
    // e<8 -> acc0 row e (batch b = e + 8*half), e>=8 -> acc1 row e-8 (b += 16)
    float V1[16], A10[16], A11[16], I1[16];
#pragma unroll
    for (int e = 0; e < 16; ++e) { V1[e]=0.f; A10[e]=0.f; A11[e]=0.f; I1[e]=0.f; }

    __syncthreads();

    const __bf16* a0p0 = o0 + (size_t)(l16)      * N0;   // A rows 0..15  (GEMM1)
    const __bf16* a1p0 = o0 + (size_t)(16 + l16) * N0;   // A rows 16..31 (GEMM1)
    const __bf16* a0p1 = o1 + (size_t)(l16)      * N1;   // A rows 0..15  (GEMM2)
    const __bf16* a1p1 = o1 + (size_t)(16 + l16) * N1;   // A rows 16..31 (GEMM2)

    const v8f vzero = {};

    for (int t = 0; t < T_STEPS; ++t) {
        const float* xt = x + (size_t)t * (BATCH * N0);
        // x_t is consumed at the *end* of this iteration; GEMMs hide latency.
        __builtin_prefetch(xt + tid * 8, 0, 0);

        // Opaque uniform bases (SGPR pairs): stops LICM from hoisting all 24
        // B-fragments into (spilled) registers.  addrspace(1) cast keeps the
        // loads on the GLOBAL path (LOADcnt only).
        const __bf16* W1o = W1b;
        const __bf16* Wro = Wrb;
        asm volatile("" : "+s"(W1o), "+s"(Wro));
        gbf16* w1g = (gbf16*)W1o;
        gbf16* wrg = (gbf16*)Wro;

        // --- GEMM1: h1 = o0 @ W1^T  (o0 = S0(t-1), zero at t=0) -------------
        v8f acc0 = vzero, acc1 = vzero;
#pragma unroll
        for (unsigned kc = 0; kc < N0 / 32; ++kc) {
            v16bf fa0 = load_frag_a(a0p0, kc, half);
            v16bf fa1 = load_frag_a(a1p0, kc, half);
            v16bf fb  = load_frag_b(w1g, voff1, kc);
            acc0 = __builtin_amdgcn_wmma_f32_16x16x32_bf16(false, fa0, false, fb, (short)0, acc0, false, false);
            acc1 = __builtin_amdgcn_wmma_f32_16x16x32_bf16(false, fa1, false, fb, (short)0, acc1, false, false);
        }

        // --- GLIF call 1 on layer-1 state (consumes acc0/acc1) --------------
        float* outA = out + (size_t)(2 * t) * (BATCH * N1);
#pragma unroll
        for (int e = 0; e < 16; ++e) {
            const int r  = e & 7;
            const int mt = e >> 3;
            const int b  = r + 8 * half + 16 * mt;
            float h1 = (e < 8) ? acc0[e] : acc1[e - 8];
            I1[e] = I1[e] * C_ISYN + C_DTK * h1;
            V1[e] = V1[e] * C_VDEC + C_VIN * (I1[e] + A10[e] + A11[e]);
            float Sa = glif_rate(V1[e]);
            A10[e] = A10[e] * C_A0D + C_A0M * Sa;
            A11[e] = A11[e] * C_A1D + C_A1M * Sa;
            __builtin_nontemporal_store(Sa, outA + (size_t)b * N1 + ncol);
        }

        // --- GEMM2: h2 = o1 @ Wr^T  (o1 = S_b(t-1); independent of GLIF1) ---
        acc0 = vzero; acc1 = vzero;
#pragma unroll
        for (unsigned kc = 0; kc < N1 / 32; ++kc) {
            v16bf fa0 = load_frag_a(a0p1, kc, half);
            v16bf fa1 = load_frag_a(a1p1, kc, half);
            v16bf fb  = load_frag_b(wrg, voffr, kc);
            acc0 = __builtin_amdgcn_wmma_f32_16x16x32_bf16(false, fa0, false, fb, (short)0, acc0, false, false);
            acc1 = __builtin_amdgcn_wmma_f32_16x16x32_bf16(false, fa1, false, fb, (short)0, acc1, false, false);
        }
        __syncthreads();   // B2: all GEMM reads of o0/o1 done -> safe to overwrite

        // --- GLIF call 2 (consumes acc0/acc1), publish o1 = S_b(t) ----------
        float* outB = out + (size_t)(2 * t + 1) * (BATCH * N1);
#pragma unroll
        for (int e = 0; e < 16; ++e) {
            const int r  = e & 7;
            const int mt = e >> 3;
            const int b  = r + 8 * half + 16 * mt;
            float h2 = (e < 8) ? acc0[e] : acc1[e - 8];
            I1[e] = I1[e] * C_ISYN + C_DTK * h2;
            V1[e] = V1[e] * C_VDEC + C_VIN * (I1[e] + A10[e] + A11[e]);
            float Sb = glif_rate(V1[e]);
            A10[e] = A10[e] * C_A0D + C_A0M * Sb;
            A11[e] = A11[e] * C_A1D + C_A1M * Sb;
            __builtin_nontemporal_store(Sb, outB + (size_t)b * N1 + ncol);
            o1[b * N1 + ncol] = (__bf16)Sb;       // delay-1 -> GEMM2 at t+1
        }

        // --- Layer-0 GLIF for x_t, publish o0 = S0(t) (used by GEMM1 at t+1)
#pragma unroll
        for (int k = 0; k < 8; ++k) {
            int i = tid + 1024 * k;
            float xv = xt[i];
            I0[k] = I0[k] * C_ISYN + C_DTK * xv;
            V0[k] = V0[k] * C_VDEC + C_VIN * (I0[k] + A00[k] + A01[k]);
            float S = glif_rate(V0[k]);
            A00[k] = A00[k] * C_A0D + C_A0M * S;
            A01[k] = A01[k] * C_A1D + C_A1M * S;
            o0[i]  = (__bf16)S;
        }
        __syncthreads();   // B1: o0(t)/o1(t) published for step t+1
    }
}

extern "C" void kernel_launch(void* const* d_in, const int* in_sizes, int n_in,
                              void* d_out, int out_size, void* d_ws, size_t ws_size,
                              hipStream_t stream) {
    const float* x  = (const float*)d_in[0];   // [1000,32,1,256]
    const float* W1 = (const float*)d_in[1];   // [512,256]
    const float* Wr = (const float*)d_in[2];   // [512,512]

    __bf16* W1b = (__bf16*)d_ws;                                   // 256 KB
    __bf16* Wrb = (__bf16*)((char*)d_ws + (size_t)N1 * N0 * 2);    // 512 KB

    convert_w<<<(N1 * N1 + 255) / 256, 256, 0, stream>>>(W1, Wr, W1b, Wrb);
    rsnn_persistent<<<1, 1024, 0, stream>>>(x, W1b, Wrb, (float*)d_out);
}
